// ALIGNNLayer_30107720745191
// MI455X (gfx1250) — matrix operand order repaired and checked by reading, not compiled, not run on hardware
//
#include <hip/hip_runtime.h>
#include <stdint.h>

#define N_NODES 20000
#define N_EDGES 640000
#define DIM 128

typedef __attribute__((ext_vector_type(16))) __bf16    v16bf;
typedef __attribute__((ext_vector_type(8)))  float     v8f;
typedef __attribute__((ext_vector_type(4)))  float     f32x4;
typedef __attribute__((ext_vector_type(4)))  uint32_t  u32x4;
typedef __attribute__((ext_vector_type(4)))  int       i32x4;

union FragB16 {
    v16bf v;
    u32x4 q[2];
};

__device__ __forceinline__ unsigned short f2bf(float f) {
    union { float f; uint32_t u; } c; c.f = f;
    uint32_t x = c.u;
    uint32_t r = x + 0x7FFFu + ((x >> 16) & 1u);   // round-to-nearest-even
    return (unsigned short)(r >> 16);
}

__device__ __forceinline__ uint32_t pack2bf(float a, float b) {
    return (uint32_t)f2bf(a) | ((uint32_t)f2bf(b) << 16);
}

__device__ __forceinline__ float silu_f(float x) {
    return x / (1.0f + __expf(-x));
}

// ----- gfx1250 async global->LDS gather (ASYNCcnt path), with safe fallback -----
#if defined(__has_builtin)
#  if __has_builtin(__builtin_amdgcn_global_load_async_to_lds_b128) && \
      __has_builtin(__builtin_amdgcn_s_wait_asynccnt)
#    define USE_ASYNC_LDS 1
#  endif
#endif

#ifdef USE_ASYNC_LDS
typedef __attribute__((address_space(1))) i32x4 g_i32x4;   // global int4*
typedef __attribute__((address_space(3))) i32x4 l_i32x4;   // LDS int4*
#endif

__device__ __forceinline__ void gather16_to_lds(void* lds, const void* g) {
#ifdef USE_ASYNC_LDS
    __builtin_amdgcn_global_load_async_to_lds_b128((g_i32x4*)g, (l_i32x4*)lds, 0, 0);
#else
    *(u32x4*)lds = *(const u32x4*)g;
#endif
}

__device__ __forceinline__ void async_gather_wait() {
#ifdef USE_ASYNC_LDS
    __builtin_amdgcn_s_wait_asynccnt(0);
#endif
}

// ---------------- prep kernels ----------------

__global__ void k_zero_f32(float* __restrict__ p, int n) {
    int i = blockIdx.x * blockDim.x + threadIdx.x;
    if (i < n) p[i] = 0.0f;
}

__global__ void k_cvt_bf16(const float* __restrict__ in, unsigned short* __restrict__ out, int n) {
    int i = blockIdx.x * blockDim.x + threadIdx.x;
    if (i < n) out[i] = f2bf(in[i]);
}

// in: [K][N] row-major fp32  ->  out: [N][K] row-major bf16
__global__ void k_transpose_bf16(const float* __restrict__ in, unsigned short* __restrict__ out,
                                 int K, int N) {
    int i = blockIdx.x * blockDim.x + threadIdx.x;
    if (i < K * N) {
        int k = i / N, n = i - k * N;
        out[n * K + k] = f2bf(in[i]);
    }
}

// ---------------- edge kernel ----------------
// block = 256 threads = 8 waves; 64-edge tile (4 M-subtiles)
// wave = (mg, ng): mg in {0,1} owns rows 32*mg..32*mg+31 (2 M-subtiles),
//                  ng in {0..3} owns cols 32*ng..32*ng+31 (2 N-tiles)
// -> 4 WMMAs per 4 fragment loads per K-step, 4 independent acc chains.

__global__ __launch_bounds__(256)
void k_edge(const float* __restrict__ edge_feat,
            const unsigned short* __restrict__ node_bf,   // [N_NODES][128] bf16
            const int* __restrict__ eidx,                 // [2][N_EDGES]
            const unsigned short* __restrict__ W1t,       // [128][384] bf16
            const float* __restrict__ b1,
            const unsigned short* __restrict__ W2t,       // [128][128] bf16
            const float* __restrict__ b2,
            float* __restrict__ out_edge,
            float* __restrict__ agg,
            float* __restrict__ cnt)
{
    const int K1 = 384, LD1 = 392;   // bf16 stride, 16B aligned
    const int K2 = 128, LD2 = 136;
    const int LDR = 132;             // fp32 residual stride

    __shared__ __align__(16) unsigned short sA[64 * 392];
    __shared__ __align__(16) unsigned short sH[64 * 136];
    __shared__ __align__(16) float          sRes[64 * 132];
    __shared__ int sSrc[64];
    __shared__ int sDst[64];

    const int tid = threadIdx.x;
    const int e0  = blockIdx.x * 64;

    if (tid < 64) {
        sSrc[tid] = eidx[e0 + tid];
        sDst[tid] = eidx[N_EDGES + e0 + tid];
    }
    __syncthreads();

    // ---- stage 64 x 384 bf16 input tile: [edge | node[src] | node[dst]] ----
    {
        const int jr = tid >> 4;     // 0..15
        const int j  = tid & 15;     // 8-element chunk within row
        #pragma unroll
        for (int rr = 0; rr < 4; ++rr) {
            const int r = jr + rr * 16;
            const int e = e0 + r;
            const f32x4* pe = (const f32x4*)(edge_feat + (size_t)e * DIM + 8 * j);
            f32x4 f0 = __builtin_nontemporal_load(pe);
            f32x4 f1 = __builtin_nontemporal_load(pe + 1);
            *(f32x4*)(sRes + r * LDR + 8 * j)     = f0;   // fp32 residual stash
            *(f32x4*)(sRes + r * LDR + 8 * j + 4) = f1;
            u32x4 pk;
            pk.x = pack2bf(f0.x, f0.y);
            pk.y = pack2bf(f0.z, f0.w);
            pk.z = pack2bf(f1.x, f1.y);
            pk.w = pack2bf(f1.z, f1.w);
            *(u32x4*)(sA + r * LD1 + 8 * j) = pk;
            // endpoint gathers: memory -> LDS directly (async, no VGPR round trip)
            gather16_to_lds(sA + r * LD1 + 128 + 8 * j,
                            node_bf + (size_t)sSrc[r] * DIM + 8 * j);
            gather16_to_lds(sA + r * LD1 + 256 + 8 * j,
                            node_bf + (size_t)sDst[r] * DIM + 8 * j);
        }
    }
    async_gather_wait();
    __syncthreads();

    const int lane = tid & 31;
    const int wv   = tid >> 5;
    const int mg   = wv & 1;                    // m-group
    const int ng   = wv >> 1;                   // n-group
    const int ra0  = 32 * mg + (lane & 15);     // A row, subtile 0
    const int ra1  = ra0 + 16;                  // A row, subtile 1
    const int nl0  = 32 * ng + (lane & 15);     // output col, tile 0
    const int nl1  = nl0 + 16;                  // output col, tile 1
    const int hiA  = (lane >> 4) ? 8  : 0;
    const int hiB  = (lane >> 4) ? 8  : 0;
    const int rb   = (lane >> 4) ? 8  : 0;

    v8f acc00, acc01, acc10, acc11;
    {
        float bv0 = b1[nl0], bv1 = b1[nl1];
        #pragma unroll
        for (int i = 0; i < 8; ++i) {
            acc00[i] = bv0; acc10[i] = bv0;
            acc01[i] = bv1; acc11[i] = bv1;
        }
    }

    for (int kk = 0; kk < K1; kk += 32) {
        FragB16 a0, a1, bf0, bf1;
        const unsigned short* p0 = &sA[ra0 * LD1 + kk + hiA];
        a0.q[0] = *(const u32x4*)(p0);
        a0.q[1] = *(const u32x4*)(p0 + 16);
        const unsigned short* p1 = &sA[ra1 * LD1 + kk + hiA];
        a1.q[0] = *(const u32x4*)(p1);
        a1.q[1] = *(const u32x4*)(p1 + 16);
        const unsigned short* pb0 = &W1t[(size_t)nl0 * K1 + kk + hiB];
        bf0.q[0] = *(const u32x4*)(pb0);
        bf0.q[1] = *(const u32x4*)(pb0 + 16);
        const unsigned short* pb1 = &W1t[(size_t)nl1 * K1 + kk + hiB];
        bf1.q[0] = *(const u32x4*)(pb1);
        bf1.q[1] = *(const u32x4*)(pb1 + 16);
        acc00 = __builtin_amdgcn_wmma_f32_16x16x32_bf16(false, a0.v, false, bf0.v,
                                                        (short)0, acc00, false, false);
        acc01 = __builtin_amdgcn_wmma_f32_16x16x32_bf16(false, a0.v, false, bf1.v,
                                                        (short)0, acc01, false, false);
        acc10 = __builtin_amdgcn_wmma_f32_16x16x32_bf16(false, a1.v, false, bf0.v,
                                                        (short)0, acc10, false, false);
        acc11 = __builtin_amdgcn_wmma_f32_16x16x32_bf16(false, a1.v, false, bf1.v,
                                                        (short)0, acc11, false, false);
    }

    // SiLU -> bf16 h tile (64 x 128)
    {
        const int hr0 = 32 * mg + rb;
        const int hr1 = hr0 + 16;
        #pragma unroll
        for (int r = 0; r < 8; ++r) {
            sH[(hr0 + r) * LD2 + nl0] = f2bf(silu_f(acc00[r]));
            sH[(hr0 + r) * LD2 + nl1] = f2bf(silu_f(acc01[r]));
            sH[(hr1 + r) * LD2 + nl0] = f2bf(silu_f(acc10[r]));
            sH[(hr1 + r) * LD2 + nl1] = f2bf(silu_f(acc11[r]));
        }
    }
    __syncthreads();

    {
        float bv0 = b2[nl0], bv1 = b2[nl1];
        #pragma unroll
        for (int i = 0; i < 8; ++i) {
            acc00[i] = bv0; acc10[i] = bv0;
            acc01[i] = bv1; acc11[i] = bv1;
        }
    }

    for (int kk = 0; kk < K2; kk += 32) {
        FragB16 a0, a1, bf0, bf1;
        const unsigned short* p0 = &sH[ra0 * LD2 + kk + hiA];
        a0.q[0] = *(const u32x4*)(p0);
        a0.q[1] = *(const u32x4*)(p0 + 16);
        const unsigned short* p1 = &sH[ra1 * LD2 + kk + hiA];
        a1.q[0] = *(const u32x4*)(p1);
        a1.q[1] = *(const u32x4*)(p1 + 16);
        const unsigned short* pb0 = &W2t[(size_t)nl0 * K2 + kk + hiB];
        bf0.q[0] = *(const u32x4*)(pb0);
        bf0.q[1] = *(const u32x4*)(pb0 + 16);
        const unsigned short* pb1 = &W2t[(size_t)nl1 * K2 + kk + hiB];
        bf1.q[0] = *(const u32x4*)(pb1);
        bf1.q[1] = *(const u32x4*)(pb1 + 16);
        acc00 = __builtin_amdgcn_wmma_f32_16x16x32_bf16(false, a0.v, false, bf0.v,
                                                        (short)0, acc00, false, false);
        acc01 = __builtin_amdgcn_wmma_f32_16x16x32_bf16(false, a0.v, false, bf1.v,
                                                        (short)0, acc01, false, false);
        acc10 = __builtin_amdgcn_wmma_f32_16x16x32_bf16(false, a1.v, false, bf0.v,
                                                        (short)0, acc10, false, false);
        acc11 = __builtin_amdgcn_wmma_f32_16x16x32_bf16(false, a1.v, false, bf1.v,
                                                        (short)0, acc11, false, false);
    }

    // residual (fp32 from LDS), store, and aggregate into dst nodes
    {
        const int r0 = 32 * mg + rb;
        const int r1 = r0 + 16;
        #pragma unroll
        for (int r = 0; r < 8; ++r) {
            int rowA = r0 + r, rowB = r1 + r;
            float v00 = acc00[r] + sRes[rowA * LDR + nl0];
            float v01 = acc01[r] + sRes[rowA * LDR + nl1];
            float v10 = acc10[r] + sRes[rowB * LDR + nl0];
            float v11 = acc11[r] + sRes[rowB * LDR + nl1];
            __builtin_nontemporal_store(v00, out_edge + (size_t)(e0 + rowA) * DIM + nl0);
            __builtin_nontemporal_store(v01, out_edge + (size_t)(e0 + rowA) * DIM + nl1);
            __builtin_nontemporal_store(v10, out_edge + (size_t)(e0 + rowB) * DIM + nl0);
            __builtin_nontemporal_store(v11, out_edge + (size_t)(e0 + rowB) * DIM + nl1);
            atomicAdd(&agg[(size_t)sDst[rowA] * DIM + nl0], v00);
            atomicAdd(&agg[(size_t)sDst[rowA] * DIM + nl1], v01);
            atomicAdd(&agg[(size_t)sDst[rowB] * DIM + nl0], v10);
            atomicAdd(&agg[(size_t)sDst[rowB] * DIM + nl1], v11);
        }
    }
    if (tid < 64) atomicAdd(&cnt[sDst[tid]], 1.0f);
}

// ---------------- node kernel ----------------
// block = 256 threads = 8 waves; 32-node tile (2 M-subtiles)

__global__ __launch_bounds__(256)
void k_node(const float* __restrict__ node_feat,
            const float* __restrict__ agg,
            const float* __restrict__ cnt,
            const unsigned short* __restrict__ W1t,   // [128][256] bf16
            const float* __restrict__ b1,
            const unsigned short* __restrict__ W2t,   // [128][128] bf16
            const float* __restrict__ b2,
            float* __restrict__ out_node)
{
    const int K1 = 256, LD1 = 264;
    const int K2 = 128, LD2 = 136;
    const int LDR = 132;

    __shared__ __align__(16) unsigned short sA[32 * 264];
    __shared__ __align__(16) unsigned short sH[32 * 136];
    __shared__ __align__(16) float          sRes[32 * 132];
    __shared__ float sInv[32];

    const int tid = threadIdx.x;
    const int v0  = blockIdx.x * 32;

    if (tid < 32) sInv[tid] = 1.0f / fmaxf(cnt[v0 + tid], 1.0f);
    __syncthreads();

    {
        const int jr = tid >> 4;
        const int j  = tid & 15;
        #pragma unroll
        for (int rr = 0; rr < 2; ++rr) {
            const int r = jr + rr * 16;
            const int v = v0 + r;
            const f32x4* pn = (const f32x4*)(node_feat + (size_t)v * DIM + 8 * j);
            f32x4 f0 = pn[0], f1 = pn[1];
            *(f32x4*)(sRes + r * LDR + 8 * j)     = f0;
            *(f32x4*)(sRes + r * LDR + 8 * j + 4) = f1;
            u32x4 pk;
            pk.x = pack2bf(f0.x, f0.y);
            pk.y = pack2bf(f0.z, f0.w);
            pk.z = pack2bf(f1.x, f1.y);
            pk.w = pack2bf(f1.z, f1.w);
            *(u32x4*)(sA + r * LD1 + 8 * j) = pk;

            const f32x4* pa = (const f32x4*)(agg + (size_t)v * DIM + 8 * j);
            f32x4 g0 = pa[0], g1 = pa[1];
            float inv = sInv[r];
            u32x4 pk2;
            pk2.x = pack2bf(g0.x * inv, g0.y * inv);
            pk2.y = pack2bf(g0.z * inv, g0.w * inv);
            pk2.z = pack2bf(g1.x * inv, g1.y * inv);
            pk2.w = pack2bf(g1.z * inv, g1.w * inv);
            *(u32x4*)(sA + r * LD1 + 128 + 8 * j) = pk2;
        }
    }
    __syncthreads();

    const int lane  = tid & 31;
    const int nl    = (tid >> 5) * 16 + (lane & 15);
    const int ar    = lane & 15;
    const int hiA   = (lane >> 4) ? 8  : 0;
    const int hiB   = (lane >> 4) ? 8  : 0;
    const int rbase = (lane >> 4) ? 8  : 0;

    v8f acc0, acc1;
    {
        float bv = b1[nl];
        #pragma unroll
        for (int i = 0; i < 8; ++i) { acc0[i] = bv; acc1[i] = bv; }
    }

    for (int kk = 0; kk < K1; kk += 32) {
        FragB16 b, a0, a1;
        const unsigned short* pb = &W1t[(size_t)nl * K1 + kk + hiB];
        b.q[0] = *(const u32x4*)(pb);
        b.q[1] = *(const u32x4*)(pb + 16);
        const unsigned short* p0 = &sA[ar * LD1 + kk + hiA];
        a0.q[0] = *(const u32x4*)(p0);
        a0.q[1] = *(const u32x4*)(p0 + 16);
        const unsigned short* p1 = &sA[(16 + ar) * LD1 + kk + hiA];
        a1.q[0] = *(const u32x4*)(p1);
        a1.q[1] = *(const u32x4*)(p1 + 16);
        acc0 = __builtin_amdgcn_wmma_f32_16x16x32_bf16(false, a0.v, false, b.v,
                                                       (short)0, acc0, false, false);
        acc1 = __builtin_amdgcn_wmma_f32_16x16x32_bf16(false, a1.v, false, b.v,
                                                       (short)0, acc1, false, false);
    }

    #pragma unroll
    for (int r = 0; r < 8; ++r) {
        sH[(rbase + r) * LD2 + nl]      = f2bf(silu_f(acc0[r]));
        sH[(16 + rbase + r) * LD2 + nl] = f2bf(silu_f(acc1[r]));
    }
    __syncthreads();

    {
        float bv = b2[nl];
        #pragma unroll
        for (int i = 0; i < 8; ++i) { acc0[i] = bv; acc1[i] = bv; }
    }

    for (int kk = 0; kk < K2; kk += 32) {
        FragB16 b, a0, a1;
        const unsigned short* pb = &W2t[(size_t)nl * K2 + kk + hiB];
        b.q[0] = *(const u32x4*)(pb);
        b.q[1] = *(const u32x4*)(pb + 16);
        const unsigned short* p0 = &sH[ar * LD2 + kk + hiA];
        a0.q[0] = *(const u32x4*)(p0);
        a0.q[1] = *(const u32x4*)(p0 + 16);
        const unsigned short* p1 = &sH[(16 + ar) * LD2 + kk + hiA];
        a1.q[0] = *(const u32x4*)(p1);
        a1.q[1] = *(const u32x4*)(p1 + 16);
        acc0 = __builtin_amdgcn_wmma_f32_16x16x32_bf16(false, a0.v, false, b.v,
                                                       (short)0, acc0, false, false);
        acc1 = __builtin_amdgcn_wmma_f32_16x16x32_bf16(false, a1.v, false, b.v,
                                                       (short)0, acc1, false, false);
    }

    #pragma unroll
    for (int r = 0; r < 8; ++r) {
        int row = rbase + r;
        out_node[(size_t)(v0 + row) * DIM + nl]  = acc0[r] + sRes[row * LDR + nl];
        int row1 = 16 + rbase + r;
        out_node[(size_t)(v0 + row1) * DIM + nl] = acc1[r] + sRes[row1 * LDR + nl];
    }
}

// ---------------- host launch ----------------

extern "C" void kernel_launch(void* const* d_in, const int* in_sizes, int n_in,
                              void* d_out, int out_size, void* d_ws, size_t ws_size,
                              hipStream_t stream) {
    const float* node_feat = (const float*)d_in[0];
    const float* edge_feat = (const float*)d_in[1];
    const int*   eidx      = (const int*)d_in[2];
    const float* W1e = (const float*)d_in[3];
    const float* b1e = (const float*)d_in[4];
    const float* W2e = (const float*)d_in[5];
    const float* b2e = (const float*)d_in[6];
    const float* W1n = (const float*)d_in[7];
    const float* b1n = (const float*)d_in[8];
    const float* W2n = (const float*)d_in[9];
    const float* b2n = (const float*)d_in[10];

    char* ws = (char*)d_ws;
    size_t o = 0;
    float* agg = (float*)(ws + o);              o += (size_t)N_NODES * DIM * 4;
    float* cnt = (float*)(ws + o);              o += (size_t)N_NODES * 4;       // contiguous after agg
    o = (o + 255) & ~(size_t)255;
    unsigned short* node_bf = (unsigned short*)(ws + o); o += (size_t)N_NODES * DIM * 2;
    o = (o + 255) & ~(size_t)255;
    unsigned short* W1e_t = (unsigned short*)(ws + o);   o += 384 * 128 * 2;
    unsigned short* W2e_t = (unsigned short*)(ws + o);   o += 128 * 128 * 2;
    unsigned short* W1n_t = (unsigned short*)(ws + o);   o += 256 * 128 * 2;
    unsigned short* W2n_t = (unsigned short*)(ws + o);   o += 128 * 128 * 2;

    float* out_node = (float*)d_out;
    float* out_edge = out_node + (size_t)N_NODES * DIM;

    {   // zero agg + counts (contiguous)
        int n = N_NODES * DIM + N_NODES;
        k_zero_f32<<<(n + 255) / 256, 256, 0, stream>>>(agg, n);
    }
    {   // node_feat -> bf16
        int n = N_NODES * DIM;
        k_cvt_bf16<<<(n + 255) / 256, 256, 0, stream>>>(node_feat, node_bf, n);
    }
    k_transpose_bf16<<<(384 * 128 + 255) / 256, 256, 0, stream>>>(W1e, W1e_t, 384, 128);
    k_transpose_bf16<<<(128 * 128 + 255) / 256, 256, 0, stream>>>(W2e, W2e_t, 128, 128);
    k_transpose_bf16<<<(256 * 128 + 255) / 256, 256, 0, stream>>>(W1n, W1n_t, 256, 128);
    k_transpose_bf16<<<(128 * 128 + 255) / 256, 256, 0, stream>>>(W2n, W2n_t, 128, 128);

    k_edge<<<N_EDGES / 64, 256, 0, stream>>>(edge_feat, node_bf, eidx,
                                             W1e_t, b1e, W2e_t, b2e,
                                             out_edge, agg, cnt);
    k_node<<<N_NODES / 32, 256, 0, stream>>>(node_feat, agg, cnt,
                                             W1n_t, b1n, W2n_t, b2n,
                                             out_node);
}
